// GPR_GNN_53523882443605
// MI455X (gfx1250) — matrix operand order repaired and checked by reading, not compiled, not run on hardware
//
#include <hip/hip_runtime.h>
#include <hip/hip_bf16.h>

typedef __attribute__((ext_vector_type(16))) __bf16 v16bf;
typedef __attribute__((ext_vector_type(8)))  __bf16 v8bf;
typedef __attribute__((ext_vector_type(8)))  float  v8f;

// ---------------------------------------------------------------------------
// One-time weight prep: W[K, Ncols] f32 -> Wt[Ncols, K] bf16 (transposed), so
// a B fragment (col = lane%16, K = kb + 16*(lane/16) + [0..15]) is 32
// contiguous bytes.
// ---------------------------------------------------------------------------
__global__ void transpose_to_bf16(const float* __restrict__ W,
                                  __bf16* __restrict__ Wt,
                                  int K, int Ncols) {
  const int i = blockIdx.x * blockDim.x + threadIdx.x;
  if (i >= K * Ncols) return;
  const int c = i / K;
  const int k = i - c * K;
  Wt[c * K + k] = (__bf16)W[k * Ncols + c];
}

// ---------------------------------------------------------------------------
// GEMM1: h1bf[N,256] = bf16(relu(x[N,512] @ W1 + b1)), via W1t bf16 [256,512].
// 16 waves/block; wave = N-tile, block = M-tile of 16 rows. A tile staged in
// LDS as bf16 with padded stride (520) so 16-lane ds_load_b128 is conflict-free.
// ---------------------------------------------------------------------------
__global__ void __launch_bounds__(512)
gemm1_wmma(const float* __restrict__ x,
           const __bf16* __restrict__ W1t,
           const float* __restrict__ b1,
           __bf16* __restrict__ h1) {
  constexpr int K = 512;
  constexpr int LDSW = K + 8;                 // pad: 260 dwords/row -> bank+4
  __shared__ __bf16 tileA[16 * LDSW];

  const int tid  = threadIdx.x;
  const int lane = tid & 31;
  const int wave = tid >> 5;                  // 0..15 -> N-tile
  const int m0   = blockIdx.x << 4;

  // Cooperative A-tile load+convert: 512 threads x 16 floats = 16x512 tile.
  {
    const int row = tid >> 5;                 // 0..15
    const int seg = tid & 31;                 // 0..31, 16 floats each
    const float4* src = (const float4*)(x + (size_t)(m0 + row) * K + seg * 16);
    __bf16* dst = &tileA[row * LDSW + seg * 16];
#pragma unroll
    for (int q = 0; q < 4; ++q) {
      float4 v = src[q];
      dst[q * 4 + 0] = (__bf16)v.x;
      dst[q * 4 + 1] = (__bf16)v.y;
      dst[q * 4 + 2] = (__bf16)v.z;
      dst[q * 4 + 3] = (__bf16)v.w;
    }
  }
  __syncthreads();

  const int half = lane >> 4;
  const int l16  = lane & 15;
  const int colB = (wave << 4) + l16;

  const __bf16* arow = &tileA[l16 * LDSW];
  const __bf16* brow = W1t + (size_t)colB * K;

  v8f acc = {};
#pragma unroll
  for (int kb = 0; kb < K; kb += 32) {
    // A 16x32 fragment: 16 contiguous bf16 at K = kb + 16*half.
    v8bf a0 = *(const v8bf*)(arow + kb + 16 * half);
    v8bf a1 = *(const v8bf*)(arow + kb + 16 * half + 8);
    v16bf a = __builtin_shufflevector(a0, a1, 0, 1, 2, 3, 4, 5, 6, 7,
                                      8, 9, 10, 11, 12, 13, 14, 15);
    // B 32x16 fragment: 16 contiguous bf16 at K = kb + 16*half.
    v16bf b = *(const v16bf*)(brow + kb + 16 * half);
    acc = __builtin_amdgcn_wmma_f32_16x16x32_bf16(
        false, a, false, b, (short)0, acc, false, false);
  }

  const float bv = b1[colB];
#pragma unroll
  for (int r = 0; r < 8; ++r) {
    float v = acc[r] + bv;
    v = v > 0.0f ? v : 0.0f;                  // relu
    h1[(size_t)(m0 + r + 8 * half) * 256 + colB] = (__bf16)v;
  }
}

// ---------------------------------------------------------------------------
// GEMM2: hA[N,64] = h1bf[N,256] @ W2 + b2 (f32 out), via W2t bf16 [64,256].
// 4 waves/block cover 64 columns; A tile (16x256 bf16) staged in LDS.
// ---------------------------------------------------------------------------
__global__ void __launch_bounds__(128)
gemm2_wmma(const __bf16* __restrict__ h1,
           const __bf16* __restrict__ W2t,
           const float* __restrict__ b2,
           float* __restrict__ hA) {
  constexpr int K = 256;
  constexpr int LDSW = K + 8;                 // 132 dwords/row -> bank+4
  __shared__ __bf16 tileA[16 * LDSW];

  const int tid  = threadIdx.x;
  const int lane = tid & 31;
  const int wave = tid >> 5;                  // 0..3 -> N-tile
  const int m0   = blockIdx.x << 4;

  // Cooperative copy: 128 threads x 32 bf16 (64B) = 16x256 bf16 tile.
  {
    const int row   = tid >> 3;               // 0..15
    const int chunk = tid & 7;                // 0..7
    const uint4* src = (const uint4*)(h1 + (size_t)(m0 + row) * K + chunk * 32);
    uint4* dst = (uint4*)&tileA[row * LDSW + chunk * 32];
#pragma unroll
    for (int q = 0; q < 4; ++q) dst[q] = src[q];
  }
  __syncthreads();

  const int half = lane >> 4;
  const int l16  = lane & 15;
  const int colB = (wave << 4) + l16;

  const __bf16* arow = &tileA[l16 * LDSW];
  const __bf16* brow = W2t + (size_t)colB * K;

  v8f acc = {};
#pragma unroll
  for (int kb = 0; kb < K; kb += 32) {
    v8bf a0 = *(const v8bf*)(arow + kb + 16 * half);
    v8bf a1 = *(const v8bf*)(arow + kb + 16 * half + 8);
    v16bf a = __builtin_shufflevector(a0, a1, 0, 1, 2, 3, 4, 5, 6, 7,
                                      8, 9, 10, 11, 12, 13, 14, 15);
    v16bf b = *(const v16bf*)(brow + kb + 16 * half);
    acc = __builtin_amdgcn_wmma_f32_16x16x32_bf16(
        false, a, false, b, (short)0, acc, false, false);
  }

  const float bv = b2[colB];
#pragma unroll
  for (int r = 0; r < 8; ++r) {
    hA[(size_t)(m0 + r + 8 * half) * 64 + colB] = acc[r] + bv;
  }
}

// ---------------------------------------------------------------------------
// SPMM scatter: dst[row[e], :] += w[e] * src[col[e], :]   (64 feats/row)
// One wave per edge; lane handles elements lane and lane+32. Destination is
// 25.6 MB -> L2-resident atomics.
// ---------------------------------------------------------------------------
__global__ void spmm_scatter(const int* __restrict__ erow,
                             const int* __restrict__ ecol,
                             const float* __restrict__ ew,
                             const float* __restrict__ src,
                             float* __restrict__ dst,
                             int nedges) {
  const int e = blockIdx.x * (blockDim.x >> 5) + (threadIdx.x >> 5);
  if (e >= nedges) return;
  const int lane = threadIdx.x & 31;
  const int r = erow[e];
  const int c = ecol[e];
  const float wt = ew[e];
  const float* s = src + (size_t)c * 64;
  float* d = dst + (size_t)r * 64;
  atomicAdd(&d[lane],      wt * s[lane]);
  atomicAdd(&d[lane + 32], wt * s[lane + 32]);
}

// acc += dw[hop] * h   (float4 vectorized)
__global__ void axpy_acc(float* __restrict__ acc,
                         const float* __restrict__ h,
                         const float* __restrict__ dw,
                         int hop, int n4) {
  const int i = blockIdx.x * blockDim.x + threadIdx.x;
  if (i >= n4) return;
  const float a = dw[hop];
  float4 hv = ((const float4*)h)[i];
  float4 av = ((float4*)acc)[i];
  av.x += a * hv.x; av.y += a * hv.y; av.z += a * hv.z; av.w += a * hv.w;
  ((float4*)acc)[i] = av;
}

// log_softmax over 64 classes: one wave32 per node, 2 values per lane.
__global__ void logsoftmax64(const float* __restrict__ acc,
                             float* __restrict__ out,
                             int nnodes) {
  const int node = blockIdx.x * (blockDim.x >> 5) + (threadIdx.x >> 5);
  if (node >= nnodes) return;
  const int lane = threadIdx.x & 31;
  const float v0 = acc[(size_t)node * 64 + lane];
  const float v1 = acc[(size_t)node * 64 + 32 + lane];
  float m = fmaxf(v0, v1);
#pragma unroll
  for (int off = 16; off > 0; off >>= 1)
    m = fmaxf(m, __shfl_xor(m, off, 32));
  float s = __expf(v0 - m) + __expf(v1 - m);
#pragma unroll
  for (int off = 16; off > 0; off >>= 1)
    s += __shfl_xor(s, off, 32);
  const float lse = m + __logf(s);
  out[(size_t)node * 64 + lane]      = v0 - lse;
  out[(size_t)node * 64 + 32 + lane] = v1 - lse;
}

extern "C" void kernel_launch(void* const* d_in, const int* in_sizes, int n_in,
                              void* d_out, int out_size, void* d_ws, size_t ws_size,
                              hipStream_t stream) {
  const float* x   = (const float*)d_in[0];   // [N, 512]
  const int*   er  = (const int*)  d_in[1];   // [E]
  const int*   ec  = (const int*)  d_in[2];   // [E]
  const float* ew  = (const float*)d_in[3];   // [E]
  const float* W1  = (const float*)d_in[4];   // [512, 256]
  const float* b1  = (const float*)d_in[5];   // [256]
  const float* W2  = (const float*)d_in[6];   // [256, 64]
  const float* b2  = (const float*)d_in[7];   // [64]
  const float* dw  = (const float*)d_in[8];   // [5]
  float* out = (float*)d_out;

  const int NFEAT = 512, NHID = 256, NCLASS = 64, NHOPS = 5;
  const int N = in_sizes[0] / NFEAT;          // 100000 (divisible by 16)
  const int E = in_sizes[1];                  // 3200000

  // Workspace layout (~128.3 MB):
  char* p = (char*)d_ws;
  __bf16* W1t  = (__bf16*)p;                   p += (size_t)NHID * NFEAT * 2;   // 256 KB
  __bf16* W2t  = (__bf16*)p;                   p += (size_t)NCLASS * NHID * 2;  // 32 KB
  __bf16* h1bf = (__bf16*)p;                   p += (size_t)N * NHID * 2;       // 51.2 MB
  float*  hA   = (float*)p;                    p += (size_t)N * NCLASS * 4;     // 25.6 MB
  float*  hB   = (float*)p;                    p += (size_t)N * NCLASS * 4;     // 25.6 MB
  float*  acc  = (float*)p;                                                     // 25.6 MB
  const size_t HB = (size_t)N * NCLASS * sizeof(float);

  // One-time weight transpose+convert (tiny).
  transpose_to_bf16<<<(NHID * NFEAT + 255) / 256, 256, 0, stream>>>(W1, W1t, NFEAT, NHID);
  transpose_to_bf16<<<(NCLASS * NHID + 255) / 256, 256, 0, stream>>>(W2, W2t, NHID, NCLASS);

  const int mtiles = N / 16;

  // GEMM1: h1bf = bf16(relu(x @ W1 + b1))
  gemm1_wmma<<<mtiles, 512, 0, stream>>>(x, W1t, b1, h1bf);

  // GEMM2: hA = h1bf @ W2 + b2
  gemm2_wmma<<<mtiles, 128, 0, stream>>>(h1bf, W2t, b2, hA);

  // Propagation hops with ping-pong buffers.
  hipMemsetAsync(acc, 0, HB, stream);
  float* src = hA;
  float* dst = hB;
  const int spmm_grid = (E + 7) / 8;           // 8 waves (edges) per block
  const int n4 = N * NCLASS / 4;
  const int axpy_grid = (n4 + 255) / 256;
  for (int i = 0; i < NHOPS; ++i) {
    hipMemsetAsync(dst, 0, HB, stream);
    spmm_scatter<<<spmm_grid, 256, 0, stream>>>(er, ec, ew, src, dst, E);
    axpy_acc<<<axpy_grid, 256, 0, stream>>>(acc, dst, dw, i, n4);
    float* t = src; src = dst; dst = t;
  }

  // out = log_softmax(acc, axis=1)
  logsoftmax64<<<(N + 7) / 8, 256, 0, stream>>>(acc, out, N);
}
